// LocalFeatureAggregation_25555055411640
// MI455X (gfx1250) — compile-verified, matrix operand order checked by reading, not run on hardware
//
#include <hip/hip_runtime.h>
#include <hip/hip_bf16.h>

// MI455X / gfx1250: wave32, WMMA 16x16x32 bf16 + Tensor Data Mover staging.

typedef __attribute__((ext_vector_type(16))) __bf16 bfx16;
typedef __attribute__((ext_vector_type(8)))  float  v8f;
typedef __attribute__((ext_vector_type(4)))  unsigned int u32x4;
typedef __attribute__((ext_vector_type(8)))  int   i32x8;
typedef __attribute__((ext_vector_type(4)))  int   i32x4;

#define BATCH 4
#define NPTS  8192
#define KNN   16
#define DIN   64
#define HCH   64
#define DOUT  128

#if defined(__has_builtin)
#  if __has_builtin(__builtin_amdgcn_tensor_load_to_lds)
#    define HAVE_TDM 1
#  endif
#endif
#if defined(__has_include)
#  if __has_include(<hip/amd_detail/amd_gfx1250_TDM.h>)
#    define TDM_ARITY6 1   // therock-10.0 headers -> 6-arg builtin
#  endif
#endif

static __device__ __forceinline__ v8f vzero() {
  v8f z;
#pragma unroll
  for (int i = 0; i < 8; ++i) z[i] = 0.0f;
  return z;
}

static __device__ __forceinline__ bfx16 bzero() {
  bfx16 z;
#pragma unroll
  for (int i = 0; i < 16; ++i) z[i] = (__bf16)0.0f;
  return z;
}

static __device__ __forceinline__ v8f wmma_bf16(bfx16 a, bfx16 b, v8f c) {
  // D = A(16x32 bf16) x B(32x16 bf16) + C(16x16 f32)
  return __builtin_amdgcn_wmma_f32_16x16x32_bf16(
      false, a, false, b, (short)0, c, false, false);
}

#if HAVE_TDM
// TDM: load a 3 x 256 f32 tile out of a row-major [3][NPTS] tensor into LDS.
// D# per CDNA5 ISA: group0 = {count, lds_addr, global_addr, type=2},
// group1 = {data_size=4B, tensor_dim0=NPTS, tensor_dim1=3, tile=256x3, dim0_stride=NPTS}.
static __device__ __forceinline__ void tdm_load_tile3x256(const float* gsrc, void* lds_dst) {
  const unsigned long long ga = (unsigned long long)(uintptr_t)gsrc;
  const unsigned int       la = (unsigned int)(uintptr_t)lds_dst;  // flat[31:0] == LDS offset
  u32x4 g0;
  g0[0] = 1u;                                                   // count=1 (valid)
  g0[1] = la;                                                   // lds_addr
  g0[2] = (unsigned int)ga;                                     // global_addr[31:0]
  g0[3] = (unsigned int)((ga >> 32) & 0x01FFFFFFu) | (2u << 30);// global_addr[56:32] | type=2
  i32x8 g1;
  g1[0] = (int)(2u << 16);                    // data_size = 4 bytes
  g1[1] = (int)(((unsigned)NPTS & 0xFFFFu) << 16);  // tensor_dim0 lo16
  g1[2] = (int)(3u << 16);                    // tensor_dim0 hi16=0 | tensor_dim1 lo16 = 3
  g1[3] = (int)(256u << 16);                  // tensor_dim1 hi16=0 | tile_dim0 = 256
  g1[4] = 3;                                  // tile_dim1 = 3 | tile_dim2 = 0
  g1[5] = NPTS;                               // tensor_dim0_stride lo32
  g1[6] = 0;                                  // stride hi | tensor_dim1_stride lo
  g1[7] = 0;
  i32x4 z4 = {0, 0, 0, 0};
#if TDM_ARITY6
  i32x8 z8 = {0, 0, 0, 0, 0, 0, 0, 0};
  __builtin_amdgcn_tensor_load_to_lds(g0, g1, z4, z4, z8, 0);
#else
  __builtin_amdgcn_tensor_load_to_lds(g0, g1, z4, z4, 0);
#endif
}
#endif

// ---------------------------------------------------------------------------
// Kernel 1: brute-force KNN (top-16 smallest squared distances).
// One thread per query; candidate tiles staged LDS-side by the TDM.
// ---------------------------------------------------------------------------
__global__ __launch_bounds__(256) void knn_topk_kernel(
    const float* __restrict__ xyz, int* __restrict__ idx) {
  const int b = blockIdx.y;
  const int q = blockIdx.x * 256 + threadIdx.x;
  const float* X = xyz + (size_t)b * 3 * NPTS;
  const float qx = X[q];
  const float qy = X[NPTS + q];
  const float qz = X[2 * NPTS + q];

  float dist[KNN];
  int   ind[KNN];
#pragma unroll
  for (int k = 0; k < KNN; ++k) { dist[k] = 3.0e38f; ind[k] = 0; }
  float worst = 3.0e38f;
  int   wslot = KNN - 1;

  __shared__ float stile[3 * 256];  // rows x,y,z of the current 256-pt tile
  const float* sx = stile;
  const float* sy = stile + 256;
  const float* sz = stile + 512;

  for (int base = 0; base < NPTS; base += 256) {
#if HAVE_TDM
    if (threadIdx.x < 32) {            // wave 0 drives the DMA
      tdm_load_tile3x256(X + base, stile);
      __builtin_amdgcn_s_wait_tensorcnt(0);
    }
#else
    stile[threadIdx.x]       = X[base + threadIdx.x];
    stile[256 + threadIdx.x] = X[NPTS + base + threadIdx.x];
    stile[512 + threadIdx.x] = X[2 * NPTS + base + threadIdx.x];
#endif
    __syncthreads();
#pragma unroll 4
    for (int t = 0; t < 256; ++t) {
      const float dx = qx - sx[t];
      const float dy = qy - sy[t];
      const float dz = qz - sz[t];
      const float d = dx * dx + dy * dy + dz * dz;
      if (d < worst) {
#pragma unroll
        for (int k = 0; k < KNN; ++k)
          if (k == wslot) { dist[k] = d; ind[k] = base + t; }
        worst = dist[0]; wslot = 0;
#pragma unroll
        for (int k = 1; k < KNN; ++k)
          if (dist[k] > worst) { worst = dist[k]; wslot = k; }
      }
    }
    __syncthreads();
  }

  int* o = idx + (size_t)(b * NPTS + q) * KNN;
#pragma unroll
  for (int k = 0; k < KNN; ++k) o[k] = ind[k];
}

// ---------------------------------------------------------------------------
// Kernel 2: shortcut GEMM  out[b,o,n] = bs[o] + sum_c ws[o,c] * feature[b,c,n]
// One wave per 16x32 output strip: A (weights) frags loaded once, reused
// across two 16x16 N-tiles -> 4 WMMAs per A-load. Fully overwrites d_out.
// ---------------------------------------------------------------------------
__global__ __launch_bounds__(256) void shortcut_gemm_kernel(
    const float* __restrict__ feat, const float* __restrict__ ws,
    const float* __restrict__ bs, float* __restrict__ out) {
  const int warp = threadIdx.x >> 5;
  const int lane = threadIdx.x & 31;
  const int half = lane >> 4;
  const int col  = lane & 15;
  const int wid  = blockIdx.x * 8 + warp;

  const int stripsPerB = (DOUT / 16) * (NPTS / 32);  // 8 * 256
  const int b  = wid / stripsPerB;
  const int r0 = wid % stripsPerB;
  const int ot = r0 / (NPTS / 32);
  const int np = r0 % (NPTS / 32);
  const int o0 = ot * 16, n0 = np * 32;

  // ISA 16-bit operand layout: lane<16 -> K in [0..7]+[16..23], lane>=16 -> +8
  const int baseLo = half ? 8 : 0;
  const int baseHi = half ? 24 : 16;
  const int arow = o0 + col;

  bfx16 a[2];
#pragma unroll
  for (int qk = 0; qk < 2; ++qk) {
    const int koff = qk * 32;
#pragma unroll
    for (int e = 0; e < 8; ++e) {
      a[qk][e]     = (__bf16)ws[arow * DIN + koff + baseLo + e];
      a[qk][8 + e] = (__bf16)ws[arow * DIN + koff + baseHi + e];
    }
  }

#pragma unroll
  for (int s = 0; s < 2; ++s) {
    const int ncol = n0 + s * 16 + col;
    v8f c = vzero();
#pragma unroll
    for (int qk = 0; qk < 2; ++qk) {
      const int koff = qk * 32;
      bfx16 bb;
#pragma unroll
      for (int e = 0; e < 8; ++e) {
        bb[e]     = (__bf16)feat[((size_t)b * DIN + koff + baseLo + e) * NPTS + ncol];
        bb[8 + e] = (__bf16)feat[((size_t)b * DIN + koff + baseHi + e) * NPTS + ncol];
      }
      c = wmma_bf16(a[qk], bb, c);
    }
#pragma unroll
    for (int r = 0; r < 8; ++r) {
      const int o = o0 + r + 8 * half;  // C layout: VGPR r holds rows r / r+8
      out[((size_t)b * DOUT + o) * NPTS + ncol] = c[r] + bs[o];
    }
  }
}

// ---------------------------------------------------------------------------
// Kernel 3: per-point MLP (10->64->64->64) over 16 neighbors + max-pool.
// TWO points per wave so every weight B-frag build feeds 2 WMMAs; inter-layer
// activations kept as bf16 in LDS (contiguous 16B runs -> ds_load_b128).
// ---------------------------------------------------------------------------
__global__ __launch_bounds__(256) void lfa_mlp_kernel(
    const float* __restrict__ xyz, const int* __restrict__ idx,
    const float* __restrict__ w1, const float* __restrict__ b1,
    const float* __restrict__ w2, const float* __restrict__ b2,
    const float* __restrict__ w3, const float* __restrict__ b3,
    float* __restrict__ out) {
  // Weights staged as B-operand layout [K][out_ch] in bf16.
  __shared__ __bf16 W1[32 * HCH];           // K padded 10 -> 32
  __shared__ __bf16 W2[64 * HCH];
  __shared__ __bf16 W3[64 * HCH];
  __shared__ float  SB1[HCH], SB2[HCH], SB3[HCH];
  __shared__ __bf16 SCB[8][2][16 * HCH];    // per-wave, per-point activation scratch

  for (int i = threadIdx.x; i < 32 * HCH; i += 256) {
    const int k = i >> 6, ch = i & 63;
    W1[i] = (__bf16)((k < 10) ? w1[ch * 10 + k] : 0.0f);
  }
  for (int i = threadIdx.x; i < 64 * HCH; i += 256) {
    const int k = i >> 6, ch = i & 63;
    W2[i] = (__bf16)w2[ch * 64 + k];
    W3[i] = (__bf16)w3[ch * 64 + k];
  }
  for (int i = threadIdx.x; i < HCH; i += 256) {
    SB1[i] = b1[i]; SB2[i] = b2[i]; SB3[i] = b3[i];
  }
  __syncthreads();

  const int warp = threadIdx.x >> 5;
  const int lane = threadIdx.x & 31;
  const int half = lane >> 4;
  const int col  = lane & 15;     // neighbor row for A-frags / channel col for C
  const int wid  = blockIdx.x * 8 + warp;
  const int b  = (2 * wid) / NPTS;
  const int n0 = (2 * wid) % NPTS;   // points n0, n0+1 (never straddles a batch)

  const int baseLo = half ? 8 : 0;
  const int baseHi = half ? 24 : 16;
  __bf16* sc0 = &SCB[warp][0][0];
  __bf16* sc1 = &SCB[warp][1][0];

  // Build layer-1 A-frags: pos = [cx cy cz kx ky kz dx dy dz dsq], K 10 -> 32
  const float* X = xyz + (size_t)b * 3 * NPTS;
  bfx16 a1[2];
#pragma unroll
  for (int p = 0; p < 2; ++p) {
    const int n  = n0 + p;
    const float cx = X[n], cy = X[NPTS + n], cz = X[2 * NPTS + n];
    const int   nb = idx[(size_t)(b * NPTS + n) * KNN + col];
    const float kx = X[nb], ky = X[NPTS + nb], kz = X[2 * NPTS + nb];
    const float dx = kx - cx, dy = ky - cy, dz = kz - cz;
    const float dsq = dx * dx + dy * dy + dz * dz;
    a1[p] = bzero();
    if (half == 0) {  // K = 0..7 (16..23 zero pad)
      a1[p][0] = (__bf16)cx; a1[p][1] = (__bf16)cy; a1[p][2] = (__bf16)cz;
      a1[p][3] = (__bf16)kx; a1[p][4] = (__bf16)ky; a1[p][5] = (__bf16)kz;
      a1[p][6] = (__bf16)dx; a1[p][7] = (__bf16)dy;
    } else {          // K = 8..15 (8,9 valid), 24..31 zero pad
      a1[p][0] = (__bf16)dz; a1[p][1] = (__bf16)dsq;
    }
  }

  auto bfrag = [&](const __bf16* W, int koff, int t) {
    bfx16 r;
#pragma unroll
    for (int e = 0; e < 8; ++e) r[e]     = W[(koff + baseLo + e) * HCH + t * 16 + col];
#pragma unroll
    for (int e = 0; e < 8; ++e) r[8 + e] = W[(koff + baseHi + e) * HCH + t * 16 + col];
    return r;
  };
  auto afrag = [&](const __bf16* scb, int koff) {  // contiguous 16B runs
    bfx16 r;
#pragma unroll
    for (int e = 0; e < 8; ++e) r[e]     = scb[col * HCH + koff + baseLo + e];
#pragma unroll
    for (int e = 0; e < 8; ++e) r[8 + e] = scb[col * HCH + koff + baseHi + e];
    return r;
  };

  // ---- layer 1: [16,10] x [10,64] for both points ----
#pragma unroll
  for (int t = 0; t < 4; ++t) {
    const bfx16 bf = bfrag(W1, 0, t);
    const float bv = SB1[t * 16 + col];
    v8f c0 = wmma_bf16(a1[0], bf, vzero());
    v8f c1 = wmma_bf16(a1[1], bf, vzero());
#pragma unroll
    for (int r = 0; r < 8; ++r) {
      sc0[(r + 8 * half) * HCH + t * 16 + col] = (__bf16)(c0[r] + bv);
      sc1[(r + 8 * half) * HCH + t * 16 + col] = (__bf16)(c1[r] + bv);
    }
  }

  // ---- layer 2: [16,64] x [64,64] for both points ----
  {
    const bfx16 p0a = afrag(sc0, 0), p0b = afrag(sc0, 32);
    const bfx16 p1a = afrag(sc1, 0), p1b = afrag(sc1, 32);
#pragma unroll
    for (int t = 0; t < 4; ++t) {
      const bfx16 bf0 = bfrag(W2, 0, t), bf1 = bfrag(W2, 32, t);
      const float bv = SB2[t * 16 + col];
      v8f c0 = wmma_bf16(p0a, bf0, vzero()); c0 = wmma_bf16(p0b, bf1, c0);
      v8f c1 = wmma_bf16(p1a, bf0, vzero()); c1 = wmma_bf16(p1b, bf1, c1);
#pragma unroll
      for (int r = 0; r < 8; ++r) {
        sc0[(r + 8 * half) * HCH + t * 16 + col] = (__bf16)(c0[r] + bv);
        sc1[(r + 8 * half) * HCH + t * 16 + col] = (__bf16)(c1[r] + bv);
      }
    }
  }

  // ---- layer 3 + max-pool over 16 neighbors + RMW into both output halves ----
  {
    const bfx16 p0a = afrag(sc0, 0), p0b = afrag(sc0, 32);
    const bfx16 p1a = afrag(sc1, 0), p1b = afrag(sc1, 32);
#pragma unroll
    for (int t = 0; t < 4; ++t) {
      const bfx16 bf0 = bfrag(W3, 0, t), bf1 = bfrag(W3, 32, t);
      const float bv = SB3[t * 16 + col];
      v8f c0 = wmma_bf16(p0a, bf0, vzero()); c0 = wmma_bf16(p0b, bf1, c0);
      v8f c1 = wmma_bf16(p1a, bf0, vzero()); c1 = wmma_bf16(p1b, bf1, c1);
      float m0 = c0[0], m1 = c1[0];
#pragma unroll
      for (int r = 1; r < 8; ++r) { m0 = fmaxf(m0, c0[r]); m1 = fmaxf(m1, c1[r]); }
      m0 += bv;  // bias uniform over rows: add after max
      m1 += bv;
      m0 = fmaxf(m0, __shfl_xor(m0, 16, 32));  // combine half-wave row groups
      m1 = fmaxf(m1, __shfl_xor(m1, 16, 32));
      if (half == 0) {
        const size_t o = ((size_t)b * DOUT + (t * 16 + col)) * NPTS + n0;
        out[o]     += m0;  out[o + (size_t)HCH * NPTS]     += m0;
        out[o + 1] += m1;  out[o + 1 + (size_t)HCH * NPTS] += m1;
      }
    }
  }
}

extern "C" void kernel_launch(void* const* d_in, const int* in_sizes, int n_in,
                              void* d_out, int out_size, void* d_ws, size_t ws_size,
                              hipStream_t stream) {
  const float* xyz     = (const float*)d_in[0];
  const float* feature = (const float*)d_in[1];
  const float* w1 = (const float*)d_in[2];
  const float* b1 = (const float*)d_in[3];
  const float* w2 = (const float*)d_in[4];
  const float* b2 = (const float*)d_in[5];
  const float* w3 = (const float*)d_in[6];
  const float* b3 = (const float*)d_in[7];
  const float* ws = (const float*)d_in[8];
  const float* bs = (const float*)d_in[9];
  float* out = (float*)d_out;
  int*   idx = (int*)d_ws;  // [B, N, 16] neighbor indices (2 MB scratch)

  // 1) KNN indices (TDM-staged candidate tiles)
  dim3 g1(NPTS / 256, BATCH);
  knn_topk_kernel<<<g1, 256, 0, stream>>>(xyz, idx);

  // 2) shortcut GEMM (fully overwrites d_out; replay-safe)
  const int scWaves = BATCH * (DOUT / 16) * (NPTS / 32);
  shortcut_gemm_kernel<<<scWaves / 8, 256, 0, stream>>>(feature, ws, bs, out);

  // 3) per-point MLP + max-pool (2 points per wave), accumulated into d_out
  const int mlpWaves = BATCH * NPTS / 2;
  lfa_mlp_kernel<<<mlpWaves / 8, 256, 0, stream>>>(xyz, idx, w1, b1, w2, b2, w3, b3, out);
}
